// LinearTer_87230785782560
// MI455X (gfx1250) — matrix operand order repaired and checked by reading, not compiled, not run on hardware
//
#include <hip/hip_runtime.h>
#include <cstdint>
#include <cstddef>

typedef __attribute__((ext_vector_type(16))) _Float16 v16h;
typedef __attribute__((ext_vector_type(8)))  _Float16 v8h;
typedef __attribute__((ext_vector_type(8)))  float    v8f;
typedef int v4i __attribute__((vector_size(16)));   // matches builtin param pointee

#define TOKENS 8192
#define NOUT   4096
#define KIN    4096

#define BM 128
#define BN 128
#define BK 64
#define LDSK (BK + 8)   // 72 halves -> 144B row pitch (16B multiple, spreads LDS banks)

// ---------------------------------------------------------------------------
// gfx1250 async global->LDS copy (ASYNCcnt-tracked), with sync fallback.
// ---------------------------------------------------------------------------
#if defined(__gfx1250__) && __has_builtin(__builtin_amdgcn_global_load_async_to_lds_b128)
#define TER_ASYNC_LDS 1
#endif

__device__ __forceinline__ void copy16_to_lds(_Float16* lds_dst, const _Float16* gsrc) {
#if defined(TER_ASYNC_LDS)
    __builtin_amdgcn_global_load_async_to_lds_b128(
        (__attribute__((address_space(1))) v4i*)gsrc,
        (__attribute__((address_space(3))) v4i*)lds_dst,
        0, 0);
#else
    *(uint4*)lds_dst = *(const uint4*)gsrc;
#endif
}

__device__ __forceinline__ void wait_lds_copies() {
#if defined(TER_ASYNC_LDS)
#if __has_builtin(__builtin_amdgcn_s_wait_asynccnt)
    __builtin_amdgcn_s_wait_asynccnt(0);
#else
    asm volatile("s_wait_asynccnt 0" ::: "memory");
#endif
#endif
}

// ---------------------------------------------------------------------------
// Phase 1: precision conversion into workspace
// ---------------------------------------------------------------------------
__global__ __launch_bounds__(256) void cvt_x_f16(const float* __restrict__ src,
                                                 _Float16* __restrict__ dst) {
    size_t i = ((size_t)blockIdx.x * 256u + threadIdx.x) * 8u;
    float4 f0 = *(const float4*)(src + i);
    float4 f1 = *(const float4*)(src + i + 4);
    v8h o;
    o[0] = (_Float16)f0.x; o[1] = (_Float16)f0.y; o[2] = (_Float16)f0.z; o[3] = (_Float16)f0.w;
    o[4] = (_Float16)f1.x; o[5] = (_Float16)f1.y; o[6] = (_Float16)f1.z; o[7] = (_Float16)f1.w;
    *(v8h*)(dst + i) = o;
}

__device__ __forceinline__ float tern(float w) {
    // jnp.round(jnp.clip(w,-1,1)) : clamp then round-to-nearest-even
    return rintf(fminf(fmaxf(w, -1.0f), 1.0f));
}

__global__ __launch_bounds__(256) void cvt_w_ter_f16(const float* __restrict__ src,
                                                     _Float16* __restrict__ dst) {
    size_t i = ((size_t)blockIdx.x * 256u + threadIdx.x) * 8u;
    float4 f0 = *(const float4*)(src + i);
    float4 f1 = *(const float4*)(src + i + 4);
    v8h o;
    o[0] = (_Float16)tern(f0.x); o[1] = (_Float16)tern(f0.y);
    o[2] = (_Float16)tern(f0.z); o[3] = (_Float16)tern(f0.w);
    o[4] = (_Float16)tern(f1.x); o[5] = (_Float16)tern(f1.y);
    o[6] = (_Float16)tern(f1.z); o[7] = (_Float16)tern(f1.w);
    *(v8h*)(dst + i) = o;
}

// ---------------------------------------------------------------------------
// Phase 2: WMMA GEMM. C[t,o] = sum_k A16[t,k] * W16[o,k] + bias[o]
//   A = activations (M=8192 x K=4096), B = W16^T (K x N), both stored with
//   contiguous K per row, so A- and B-tiles are identical row/K-slice copies.
// ---------------------------------------------------------------------------
union Frag { v16h v; v8h h[2]; };

__global__ __launch_bounds__(256) void wmma_ter_gemm(
    const _Float16* __restrict__ A,   // [TOKENS][KIN]
    const _Float16* __restrict__ W,   // [NOUT][KIN] (ternary, exact in f16)
    const float*    __restrict__ bias,
    float*          __restrict__ out) // [TOKENS][NOUT]
{
    __shared__ _Float16 sA[2][BM * LDSK];
    __shared__ _Float16 sB[2][BN * LDSK];

    const int tid  = threadIdx.x;
    const int lane = tid & 31;
    const int wave = tid >> 5;   // 8 waves (wave32)
    const int wm   = wave >> 1;  // 0..3 -> 32-row slice
    const int wn   = wave & 1;   // 0..1 -> 64-col slice

    const int tileM = blockIdx.y * BM;
    const int tileN = blockIdx.x * BN;

    // Cooperative tile loader: 256 threads x 4 x 16B per matrix per stage.
    const int lrow   = tid >> 3;        // 0..31
    const int lcol   = (tid & 7) * 8;   // half-offset in row: 0..56
    const int ldsOff = lrow * LDSK + lcol;
    const _Float16* gA = A + (size_t)(tileM + lrow) * KIN + lcol;
    const _Float16* gW = W + (size_t)(tileN + lrow) * KIN + lcol;

    v8f acc[2][4] = {};

    const int frow = lane & 15;  // row (A) / col (B) within 16x16 frag
    const int kgrp = lane >> 4;  // lane-group K offset selector

    const int KT = KIN / BK;     // 64 stages

    // prologue: stage 0 into buffer 0
#pragma unroll
    for (int i = 0; i < 4; ++i) {
        copy16_to_lds(&sA[0][ldsOff + i * 32 * LDSK], gA + (size_t)i * 32 * KIN);
        copy16_to_lds(&sB[0][ldsOff + i * 32 * LDSK], gW + (size_t)i * 32 * KIN);
    }

    for (int kt = 0; kt < KT; ++kt) {
        const int b = kt & 1;
        wait_lds_copies();
        __syncthreads();   // stage kt resident & all waves done with buf b^1 reads

        if (kt + 1 < KT) {
            const size_t gk = (size_t)(kt + 1) * BK;
#pragma unroll
            for (int i = 0; i < 4; ++i) {
                copy16_to_lds(&sA[b ^ 1][ldsOff + i * 32 * LDSK], gA + gk + (size_t)i * 32 * KIN);
                copy16_to_lds(&sB[b ^ 1][ldsOff + i * 32 * LDSK], gW + gk + (size_t)i * 32 * KIN);
            }
        }

#pragma unroll
        for (int ks = 0; ks < 2; ++ks) {
            const int k0 = ks * 32;
            Frag a[2], bf[4];
            // A frag: row=frow, K in [k0+kgrp*8, +8) and [k0+kgrp*8+16, +8)
#pragma unroll
            for (int mi = 0; mi < 2; ++mi) {
                const _Float16* p = &sA[b][(wm * 32 + mi * 16 + frow) * LDSK + k0 + kgrp * 8];
                a[mi].h[0] = *(const v8h*)p;
                a[mi].h[1] = *(const v8h*)(p + 16);
            }
            // B frag: col=frow, K in [k0+kgrp*8, +8) and [k0+kgrp*8+16, +8)
#pragma unroll
            for (int ni = 0; ni < 4; ++ni) {
                const _Float16* p = &sB[b][(wn * 64 + ni * 16 + frow) * LDSK + k0 + kgrp * 8];
                bf[ni].h[0] = *(const v8h*)p;
                bf[ni].h[1] = *(const v8h*)(p + 16);
            }
#pragma unroll
            for (int mi = 0; mi < 2; ++mi)
#pragma unroll
                for (int ni = 0; ni < 4; ++ni)
                    acc[mi][ni] = __builtin_amdgcn_wmma_f32_16x16x32_f16(
                        false, a[mi].v, false, bf[ni].v,
                        (short)0, acc[mi][ni], false, false);
        }
        __syncthreads();   // all reads of buf b done before it is refilled
    }

    // Epilogue: C layout -> VGPR r holds row r + kgrp*8, col = frow
#pragma unroll
    for (int ni = 0; ni < 4; ++ni) {
        const int col = tileN + wn * 64 + ni * 16 + frow;
        const float bv = bias[col];
#pragma unroll
        for (int mi = 0; mi < 2; ++mi) {
            const int r0 = tileM + wm * 32 + mi * 16 + kgrp * 8;
#pragma unroll
            for (int r = 0; r < 8; ++r)
                out[(size_t)(r0 + r) * NOUT + col] = acc[mi][ni][r] + bv;
        }
    }
}

// ---------------------------------------------------------------------------
// Safety fallback if workspace is too small: direct f32 dot products.
// ---------------------------------------------------------------------------
__global__ __launch_bounds__(256) void naive_ter(const float* __restrict__ x,
                                                 const float* __restrict__ w,
                                                 const float* __restrict__ bias,
                                                 float* __restrict__ out) {
    size_t idx = (size_t)blockIdx.x * 256u + threadIdx.x;
    size_t t = idx / NOUT, o = idx % NOUT;
    const float* xr = x + t * KIN;
    const float* wr = w + o * KIN;
    float acc = bias[o];
    for (int k = 0; k < KIN; ++k)
        acc += xr[k] * tern(wr[k]);
    out[idx] = acc;
}

// ---------------------------------------------------------------------------
extern "C" void kernel_launch(void* const* d_in, const int* in_sizes, int n_in,
                              void* d_out, int out_size, void* d_ws, size_t ws_size,
                              hipStream_t stream) {
    const float* x    = (const float*)d_in[0];
    const float* w    = (const float*)d_in[1];
    const float* bias = (const float*)d_in[2];
    float* out        = (float*)d_out;

    const size_t nx = (size_t)TOKENS * KIN;   // 33.5M
    const size_t nw = (size_t)NOUT * KIN;     // 16.8M
    const size_t need = (nx + nw) * sizeof(_Float16);  // ~100.7 MB

    if (ws_size < need) {
        const size_t total = (size_t)TOKENS * NOUT;
        naive_ter<<<dim3((unsigned)(total / 256)), dim3(256), 0, stream>>>(x, w, bias, out);
        return;
    }

    _Float16* x16 = (_Float16*)d_ws;
    _Float16* w16 = x16 + nx;

    cvt_x_f16    <<<dim3((unsigned)(nx / (8 * 256))), dim3(256), 0, stream>>>(x, x16);
    cvt_w_ter_f16<<<dim3((unsigned)(nw / (8 * 256))), dim3(256), 0, stream>>>(w, w16);

    dim3 grid(NOUT / BN, TOKENS / BM);  // (32, 64)
    wmma_ter_gemm<<<grid, dim3(256), 0, stream>>>(x16, w16, bias, out);
}
